// HybridAttention_47802986004903
// MI455X (gfx1250) — compile-verified, matrix operand order checked by reading, not run on hardware
//
#include <hip/hip_runtime.h>

// ---------------------------------------------------------------- types
typedef __attribute__((ext_vector_type(16))) __bf16 v16bf;
typedef __attribute__((ext_vector_type(8)))  float  v8f;

union FragB { v16bf v; unsigned int u[8]; };
union U16x8 { uint4 q; unsigned short us[8]; };

__device__ __forceinline__ unsigned short f2bf(float f) {
    unsigned int u = __builtin_bit_cast(unsigned int, f);
    unsigned int r = u + 0x7FFFu + ((u >> 16) & 1u);
    return (unsigned short)(r >> 16);
}
__device__ __forceinline__ float bf2f(unsigned short h) {
    unsigned int u = ((unsigned int)h) << 16;
    return __builtin_bit_cast(float, u);
}

// ---------------------------------------------------------------- dims
#define Dm   768
#define Hh   12
#define Dh   64
#define Bb   2
#define Ss   1024
#define ROWS (Bb * Ss)      // 2048
#define NCAT (6 * Dm)       // 4608  (qf kf vl qo ko vo)
#define GC   (3 * Dm)       // 2304  (lin | win | xn)

// ---------------------------------------------------------------- generic bf16 WMMA GEMM
// C(M,N) = epilogue( A(M,K) * B(K,N) + bias ), A/B bf16.
// A strides generic (one of As_k/As_m must be 1); B assumed Bs_n == 1.
#define BM 128
#define BN 128
#define BK 32
#define LDA 40          // 80B row pitch: 16B-aligned rows, conflict-free frag reads
#define LDB 40

enum { MODE_NONE = 0, MODE_PROJ = 1, MODE_ROWSCALE = 2, MODE_RELU = 3,
       MODE_GATE = 4, MODE_RESID = 5 };

struct GemmP {
    const unsigned short* A;  long long As_m, As_k, Abat;
    const unsigned short* B;  long long Bs_k, Bbat;      // Bs_n == 1
    const float* bias;
    float*          outF; long long ldF,  Fbat;
    unsigned short* outB; long long ldBo, Bobat;
    const float* rowscale; long long rsBat;
    const float* linF; const float* winF; const float* resid;
    int M, N, K, mode, eluThresh;
};

__global__ __launch_bounds__(256) void gemm_bf16(GemmP p) {
    __shared__ __align__(16) unsigned short Asl[BM][LDA];
    __shared__ __align__(16) unsigned short Bsl[BN][LDB];

    const int tid  = threadIdx.x;
    const int lane = tid & 31, wave = tid >> 5;
    const int wR = wave & 3,  wC = wave >> 2;       // 4x2 wave grid, 32x64 per wave
    const int l15 = lane & 15, lhi = lane >> 4;
    const int bz = blockIdx.z;

    const unsigned short* A  = p.A + (long long)bz * p.Abat;
    const unsigned short* Bm = p.B + (long long)bz * p.Bbat;
    const int m0 = blockIdx.y * BM;
    const int n0 = blockIdx.x * BN;

    v8f zero = {0.f,0.f,0.f,0.f,0.f,0.f,0.f,0.f};
    v8f acc[2][4];
    #pragma unroll
    for (int mi = 0; mi < 2; ++mi)
        #pragma unroll
        for (int ni = 0; ni < 4; ++ni) acc[mi][ni] = zero;

    const bool aRowContig = (p.As_k == 1);

    for (int k0 = 0; k0 < p.K; k0 += BK) {
        // ---- stage A tile (BM x BK)
        if (aRowContig) {
            // contiguous along K: DMA straight into LDS (async, 2 x b128 / thread)
            const int row = tid >> 1, ks = (tid & 1) * 16;
            const unsigned short* src =
                A + (long long)(m0 + row) * p.As_m + (k0 + ks);
            unsigned long long g = (unsigned long long)(size_t)src;
            unsigned int       l = (unsigned int)(size_t)&Asl[row][ks];
            asm volatile("global_load_async_to_lds_b128 %0, %1, off"
                         :: "v"(l), "v"(g) : "memory");
            asm volatile("global_load_async_to_lds_b128 %0, %1, off offset:16"
                         :: "v"(l), "v"(g) : "memory");
        } else {
            // contiguous along M (e.g. kf^T): wide reads, scatter into LDS
            const int k = tid >> 3, moff = (tid & 7) * 16;
            const unsigned short* src =
                A + (long long)(k0 + k) * p.As_k + (m0 + moff);
            U16x8 c0, c1;
            c0.q = ((const uint4*)src)[0];
            c1.q = ((const uint4*)src)[1];
            #pragma unroll
            for (int j = 0; j < 8; ++j) {
                Asl[moff + j][k]     = c0.us[j];
                Asl[moff + 8 + j][k] = c1.us[j];
            }
        }
        // ---- stage B tile (BK x BN): contiguous along N, stored N-major
        {
            const int k = tid >> 3, noff = (tid & 7) * 16;
            const unsigned short* src =
                Bm + (long long)(k0 + k) * p.Bs_k + (n0 + noff);
            U16x8 c0, c1;
            c0.q = ((const uint4*)src)[0];
            c1.q = ((const uint4*)src)[1];
            #pragma unroll
            for (int j = 0; j < 8; ++j) {
                Bsl[noff + j][k]     = c0.us[j];
                Bsl[noff + 8 + j][k] = c1.us[j];
            }
        }
        asm volatile("s_wait_asynccnt 0x0" ::: "memory");
        __syncthreads();

        FragB af[2], bfr[4];
        #pragma unroll
        for (int mi = 0; mi < 2; ++mi) {
            const int m = wR * 32 + mi * 16 + l15;
            #pragma unroll
            for (int r = 0; r < 8; ++r) {
                const int kb = ((r < 4) ? (2 * r) : (16 + 2 * (r - 4))) + lhi * 8;
                af[mi].u[r] = *(const unsigned int*)&Asl[m][kb];
            }
        }
        #pragma unroll
        for (int ni = 0; ni < 4; ++ni) {
            const int n = wC * 64 + ni * 16 + l15;
            #pragma unroll
            for (int r = 0; r < 8; ++r) {
                const int kb = ((r < 4) ? (2 * r) : (16 + 2 * (r - 4))) + lhi * 8;
                bfr[ni].u[r] = *(const unsigned int*)&Bsl[n][kb];
            }
        }
        #pragma unroll
        for (int mi = 0; mi < 2; ++mi)
            #pragma unroll
            for (int ni = 0; ni < 4; ++ni)
                acc[mi][ni] = __builtin_amdgcn_wmma_f32_16x16x32_bf16(
                    false, af[mi].v, false, bfr[ni].v,
                    (short)0, acc[mi][ni], false, false);
        __syncthreads();
    }

    // ---- epilogue
    #pragma unroll
    for (int mi = 0; mi < 2; ++mi) {
        #pragma unroll
        for (int ni = 0; ni < 4; ++ni) {
            #pragma unroll
            for (int i = 0; i < 8; ++i) {
                const int gm = m0 + wR * 32 + mi * 16 + lhi * 8 + i;
                const int gn = n0 + wC * 64 + ni * 16 + l15;
                float v = acc[mi][ni][i];
                if (p.bias) v += p.bias[gn];
                switch (p.mode) {
                    case MODE_PROJ:
                        if (gn < p.eluThresh) v = (v > 0.f) ? (v + 1.f) : __expf(v);
                        break;
                    case MODE_ROWSCALE:
                        v *= p.rowscale[p.rsBat * bz + gm];
                        break;
                    case MODE_RELU:
                        v = fmaxf(v, 0.f);
                        break;
                    case MODE_GATE: {
                        const float g  = 1.f / (1.f + __expf(-v));
                        const long long fi = (long long)gm * p.N + gn;
                        v = g * p.linF[fi] + (1.f - g) * p.winF[fi];
                        break;
                    }
                    case MODE_RESID:
                        v += p.resid[(long long)gm * p.N + gn];
                        break;
                    default: break;
                }
                if (p.outF)
                    p.outF[(long long)bz * p.Fbat + (long long)gm * p.ldF + gn] = v;
                if (p.outB)
                    p.outB[(long long)bz * p.Bobat + (long long)gm * p.ldBo + gn] = f2bf(v);
            }
        }
    }
}

// ---------------------------------------------------------------- LayerNorm -> bf16 slab of gcat
__global__ __launch_bounds__(256) void ln_kernel(const float* __restrict__ x,
                                                 const float* __restrict__ g,
                                                 const float* __restrict__ b,
                                                 unsigned short* __restrict__ gcat) {
    const int row = blockIdx.x;
    const int tid = threadIdx.x;
    const float* xr = x + (long long)row * Dm;
    __shared__ float s1[256], s2[256];
    const float a0 = xr[tid], a1 = xr[tid + 256], a2 = xr[tid + 512];
    s1[tid] = a0 + a1 + a2;
    s2[tid] = a0 * a0 + a1 * a1 + a2 * a2;
    __syncthreads();
    for (int st = 128; st > 0; st >>= 1) {
        if (tid < st) { s1[tid] += s1[tid + st]; s2[tid] += s2[tid + st]; }
        __syncthreads();
    }
    const float mean = s1[0] * (1.f / Dm);
    const float var  = s2[0] * (1.f / Dm) - mean * mean;
    const float inv  = rsqrtf(var + 1e-5f);
    unsigned short* dst = gcat + (long long)row * GC + 2 * Dm;
    #pragma unroll
    for (int r = 0; r < 3; ++r) {
        const int j = tid + r * 256;
        dst[j] = f2bf((xr[j] - mean) * inv * g[j] + b[j]);
    }
}

// ---------------------------------------------------------------- ksum over sequence
__global__ void ksum_kernel(const unsigned short* __restrict__ proj, float* __restrict__ ksum) {
    const int t = blockIdx.x * blockDim.x + threadIdx.x;
    if (t >= Bb * Dm) return;
    const int b = t / Dm, d = t % Dm;
    const unsigned short* p = proj + (long long)b * Ss * NCAT + Dm + d;
    float s = 0.f;
    for (int i = 0; i < Ss; ++i) s += bf2f(p[(long long)i * NCAT]);
    ksum[t] = s;
}

// ---------------------------------------------------------------- rowscale = 1/(qf.ksum + 1e-6)
__global__ void rowscale_kernel(const unsigned short* __restrict__ proj,
                                const float* __restrict__ ksum,
                                float* __restrict__ rs) {
    const int wave = (blockIdx.x * blockDim.x + threadIdx.x) >> 5;
    const int lane = threadIdx.x & 31;
    if (wave >= ROWS) return;
    const int b = wave >> 10;
    const unsigned short* q = proj + (long long)wave * NCAT;   // qf row
    const float* ks = ksum + b * Dm;
    float s = 0.f;
    for (int d = lane; d < Dm; d += 32) s += bf2f(q[d]) * ks[d];
    for (int o = 16; o > 0; o >>= 1) s += __shfl_xor(s, o, 32);
    if (lane == 0) rs[wave] = 1.f / (s + 1e-6f);
}

// ---------------------------------------------------------------- sliding-window attention
__global__ __launch_bounds__(256) void win_kernel(const unsigned short* __restrict__ proj,
                                                  float* __restrict__ winF,
                                                  unsigned short* __restrict__ gcat) {
    const int wv = threadIdx.x >> 5, lane = threadIdx.x & 31;
    const int task = blockIdx.x * 8 + wv;            // b*S*H + s*H + h
    const int h = task % Hh;
    const int srow = task / Hh;
    const int s = srow % Ss, b = srow / Ss;
    const long long rowBase = (long long)(b * Ss + s) * NCAT;

    __shared__ float qsh[8][64];
    __shared__ float psh[8][66];

    const unsigned short* qp = proj + rowBase + 3 * Dm + h * Dh;   // qo
    qsh[wv][lane * 2]     = bf2f(qp[lane * 2]);
    qsh[wv][lane * 2 + 1] = bf2f(qp[lane * 2 + 1]);
    __syncthreads();

    float sc[3];
    float mymax = -__builtin_inff();
    int cnt = 0;
    for (int w = lane; w < 65; w += 32) {
        const int idx = s - 32 + w;
        float score;
        if (idx >= 0 && idx < Ss) {
            const unsigned short* kp =
                proj + (long long)(b * Ss + idx) * NCAT + 4 * Dm + h * Dh;  // ko
            float a = 0.f;
            #pragma unroll 8
            for (int d = 0; d < Dh; ++d) a += qsh[wv][d] * bf2f(kp[d]);
            score = a * 0.125f;                         // 1/sqrt(64)
        } else {
            score = -__builtin_inff();
        }
        sc[cnt++] = score;
        mymax = fmaxf(mymax, score);
    }
    for (int o = 16; o > 0; o >>= 1) mymax = fmaxf(mymax, __shfl_xor(mymax, o, 32));
    float mysum = 0.f; cnt = 0;
    for (int w = lane; w < 65; w += 32) {
        const float e = __expf(sc[cnt] - mymax);
        sc[cnt++] = e; mysum += e;
    }
    for (int o = 16; o > 0; o >>= 1) mysum += __shfl_xor(mysum, o, 32);
    const float inv = 1.f / mysum;
    cnt = 0;
    for (int w = lane; w < 65; w += 32) psh[wv][w] = sc[cnt++] * inv;
    __syncthreads();

    const int d0 = lane * 2;
    float o0 = 0.f, o1 = 0.f;
    for (int w = 0; w < 65; ++w) {
        const float pw = psh[wv][w];
        int idx = s - 32 + w;
        idx = idx < 0 ? 0 : (idx > Ss - 1 ? Ss - 1 : idx);
        const unsigned short* vp =
            proj + (long long)(b * Ss + idx) * NCAT + 5 * Dm + h * Dh;      // vo
        o0 += pw * bf2f(vp[d0]);
        o1 += pw * bf2f(vp[d0 + 1]);
    }
    const long long orow = (long long)(b * Ss + s);
    winF[orow * Dm + h * Dh + d0]     = o0;
    winF[orow * Dm + h * Dh + d0 + 1] = o1;
    gcat[orow * GC + Dm + h * Dh + d0]     = f2bf(o0);
    gcat[orow * GC + Dm + h * Dh + d0 + 1] = f2bf(o1);
}

// ---------------------------------------------------------------- small helpers
__global__ void convert_bf16(unsigned short* dst, const float* src,
                             int rows, int cols, int dstride, int doff) {
    const int t = blockIdx.x * 256 + threadIdx.x;
    if (t >= rows * cols) return;
    const int r = t / cols, c = t % cols;
    dst[(long long)r * dstride + doff + c] = f2bf(src[t]);
}

__global__ void pack_bias(float* bcat, const float* b0, const float* b1, const float* b2,
                          const float* b3, const float* b4, const float* b5) {
    const int t = blockIdx.x * 256 + threadIdx.x;
    if (t >= NCAT) return;
    const int r = t / Dm, j = t % Dm;
    const float* p = (r == 0) ? b0 : (r == 1) ? b1 : (r == 2) ? b2
                  : (r == 3) ? b3 : (r == 4) ? b4 : b5;
    bcat[t] = p[j];
}

// ---------------------------------------------------------------- host side
extern "C" void kernel_launch(void* const* d_in, const int* in_sizes, int n_in,
                              void* d_out, int out_size, void* d_ws, size_t ws_size,
                              hipStream_t stream) {
    (void)in_sizes; (void)n_in; (void)out_size; (void)ws_size;
    const float* x   = (const float*)d_in[0];
    const float* Wp[6] = { (const float*)d_in[1], (const float*)d_in[3],
                           (const float*)d_in[5], (const float*)d_in[7],
                           (const float*)d_in[9], (const float*)d_in[11] };
    const float* bp[6] = { (const float*)d_in[2], (const float*)d_in[4],
                           (const float*)d_in[6], (const float*)d_in[8],
                           (const float*)d_in[10], (const float*)d_in[12] };
    const float* W_out = (const float*)d_in[13];
    const float* b_out = (const float*)d_in[14];
    const float* W_g1  = (const float*)d_in[15];
    const float* b_g1  = (const float*)d_in[16];
    const float* W_g2  = (const float*)d_in[17];
    const float* b_g2  = (const float*)d_in[18];
    const float* gamma = (const float*)d_in[19];
    const float* beta  = (const float*)d_in[20];

    char* ws = (char*)d_ws;
    auto alloc = [&](size_t bytes) -> char* {
        char* p = ws; ws += (bytes + 255) & ~(size_t)255; return p;
    };
    unsigned short* gcat  = (unsigned short*)alloc((size_t)ROWS * GC * 2);
    unsigned short* proj  = (unsigned short*)alloc((size_t)ROWS * NCAT * 2);
    unsigned short* wcat  = (unsigned short*)alloc((size_t)Dm * NCAT * 2);
    unsigned short* wg1b  = (unsigned short*)alloc((size_t)GC * Dm * 2);
    unsigned short* wg2b  = (unsigned short*)alloc((size_t)Dm * Dm * 2);
    unsigned short* woutb = (unsigned short*)alloc((size_t)Dm * Dm * 2);
    float*          bcat  = (float*)alloc((size_t)NCAT * 4);
    float*          ksum  = (float*)alloc((size_t)Bb * Dm * 4);
    unsigned short* kvb   = (unsigned short*)alloc((size_t)Bb * Dm * Dm * 2);
    float*          rs    = (float*)alloc((size_t)ROWS * 4);
    float*          linF  = (float*)alloc((size_t)ROWS * Dm * 4);
    float*          winF  = (float*)alloc((size_t)ROWS * Dm * 4);
    unsigned short* g1b   = (unsigned short*)alloc((size_t)ROWS * Dm * 2);
    unsigned short* fused = (unsigned short*)alloc((size_t)ROWS * Dm * 2);

    // ---- weights -> bf16
    for (int i = 0; i < 6; ++i)
        convert_bf16<<<(Dm * Dm + 255) / 256, 256, 0, stream>>>(wcat, Wp[i], Dm, Dm, NCAT, i * Dm);
    convert_bf16<<<(GC * Dm + 255) / 256, 256, 0, stream>>>(wg1b, W_g1, GC, Dm, Dm, 0);
    convert_bf16<<<(Dm * Dm + 255) / 256, 256, 0, stream>>>(wg2b, W_g2, Dm, Dm, Dm, 0);
    convert_bf16<<<(Dm * Dm + 255) / 256, 256, 0, stream>>>(woutb, W_out, Dm, Dm, Dm, 0);
    pack_bias<<<(NCAT + 255) / 256, 256, 0, stream>>>(bcat, bp[0], bp[1], bp[2], bp[3], bp[4], bp[5]);

    // ---- LayerNorm -> gcat[:,1536:2304]
    ln_kernel<<<ROWS, 256, 0, stream>>>(x, gamma, beta, gcat);

    GemmP p{};

    // ---- fused 6-way projection: xn(2048x768) @ Wcat(768x4608), elu+1 on qf,kf
    p = GemmP{};
    p.A = gcat + 2 * Dm; p.As_m = GC; p.As_k = 1; p.Abat = 0;
    p.B = wcat; p.Bs_k = NCAT; p.Bbat = 0;
    p.bias = bcat;
    p.outB = proj; p.ldBo = NCAT; p.Bobat = 0;
    p.M = ROWS; p.N = NCAT; p.K = Dm; p.mode = MODE_PROJ; p.eluThresh = 2 * Dm;
    gemm_bf16<<<dim3(NCAT / BN, ROWS / BM, 1), 256, 0, stream>>>(p);

    // ---- ksum
    ksum_kernel<<<(Bb * Dm + 255) / 256, 256, 0, stream>>>(proj, ksum);

    // ---- kv[b] = kf[b]^T (768x1024) @ vl[b] (1024x768)   (A contiguous along M)
    p = GemmP{};
    p.A = proj + Dm; p.As_m = 1; p.As_k = NCAT; p.Abat = (long long)Ss * NCAT;
    p.B = proj + 2 * Dm; p.Bs_k = NCAT; p.Bbat = (long long)Ss * NCAT;
    p.outB = kvb; p.ldBo = Dm; p.Bobat = (long long)Dm * Dm;
    p.M = Dm; p.N = Dm; p.K = Ss; p.mode = MODE_NONE;
    gemm_bf16<<<dim3(Dm / BN, Dm / BM, Bb), 256, 0, stream>>>(p);

    rowscale_kernel<<<(ROWS * 32 + 255) / 256, 256, 0, stream>>>(proj, ksum, rs);

    // ---- lin_out[b] = (qf[b] @ kv[b]) * rowscale ; also -> gcat[:,0:768] bf16
    p = GemmP{};
    p.A = proj; p.As_m = NCAT; p.As_k = 1; p.Abat = (long long)Ss * NCAT;
    p.B = kvb; p.Bs_k = Dm; p.Bbat = (long long)Dm * Dm;
    p.rowscale = rs; p.rsBat = Ss;
    p.outF = linF; p.ldF = Dm; p.Fbat = (long long)Ss * Dm;
    p.outB = gcat; p.ldBo = GC; p.Bobat = (long long)Ss * GC;
    p.M = Ss; p.N = Dm; p.K = Dm; p.mode = MODE_ROWSCALE;
    gemm_bf16<<<dim3(Dm / BN, Ss / BM, Bb), 256, 0, stream>>>(p);

    // ---- sliding-window attention -> winF + gcat[:,768:1536]
    win_kernel<<<(Bb * Ss * Hh) / 8, 256, 0, stream>>>(proj, winF, gcat);

    // ---- g1 = relu(gcat(2048x2304) @ W_g1(2304x768) + b_g1)
    p = GemmP{};
    p.A = gcat; p.As_m = GC; p.As_k = 1;
    p.B = wg1b; p.Bs_k = Dm;
    p.bias = b_g1;
    p.outB = g1b; p.ldBo = Dm;
    p.M = ROWS; p.N = Dm; p.K = GC; p.mode = MODE_RELU;
    gemm_bf16<<<dim3(Dm / BN, ROWS / BM, 1), 256, 0, stream>>>(p);

    // ---- gate = sigmoid(g1 @ W_g2 + b_g2); fused = g*lin + (1-g)*win
    p = GemmP{};
    p.A = g1b; p.As_m = Dm; p.As_k = 1;
    p.B = wg2b; p.Bs_k = Dm;
    p.bias = b_g2;
    p.linF = linF; p.winF = winF;
    p.outB = fused; p.ldBo = Dm;
    p.M = ROWS; p.N = Dm; p.K = Dm; p.mode = MODE_GATE;
    gemm_bf16<<<dim3(Dm / BN, ROWS / BM, 1), 256, 0, stream>>>(p);

    // ---- out = fused @ W_out + b_out + x
    p = GemmP{};
    p.A = fused; p.As_m = Dm; p.As_k = 1;
    p.B = woutb; p.Bs_k = Dm;
    p.bias = b_out;
    p.resid = x;
    p.outF = (float*)d_out; p.ldF = Dm;
    p.M = ROWS; p.N = Dm; p.K = Dm; p.mode = MODE_RESID;
    gemm_bf16<<<dim3(Dm / BN, ROWS / BM, 1), 256, 0, stream>>>(p);
}
